// FourierConvolution2D_25675314495958
// MI455X (gfx1250) — matrix operand order repaired and checked by reading, not compiled, not run on hardware
//
#include <hip/hip_runtime.h>
#include <hip/hip_bf16.h>

// ---------------------------------------------------------------------------
// Circular 2D convolution (FFT-conv equivalent) as implicit GEMM on WMMA bf16.
// B=16, H=W=256, C=32, F=64, KH=KW=7.
// out[b,h,w,f] = sum_{c,kh,kw} x[b,(h-kh)%256,(w-kw)%256,c] * ker[f,c,kh,kw]
//
// MI455X plan:
//  * compute-bound (210 GFLOP vs ~400 MB traffic, ~520 FLOP/B) -> bf16 WMMA
//  * C=32 == K of v_wmma_f32_16x16x32_bf16: one WMMA per tap per 16x16 tile
//  * bf16 x (64 MB) + packed weights (200 KB) both L2-resident (192 MB L2)
//  * packed weights DMA'd into LDS by the Tensor Data Mover (one
//    tensor_load_to_lds per block, TENSORcnt-tracked) -> zero VGPR traffic;
//    1024-thread workgroups (32 waves) share one LDS copy so hot-loop B
//    reads are short-latency ds_load_b128
// ---------------------------------------------------------------------------

typedef __attribute__((ext_vector_type(16))) __bf16 v16bf;
typedef __attribute__((ext_vector_type(8)))  __bf16 v8bf;
typedef __attribute__((ext_vector_type(8)))  float  v8f;

typedef __attribute__((ext_vector_type(4))) unsigned int u32x4;
typedef __attribute__((ext_vector_type(4))) int          i32x4;
typedef __attribute__((ext_vector_type(8))) int          i32x8;

#define BB 16
#define HH 256
#define WW 256
#define CC 32
#define FF 64
#define KH 7
#define KW 7

#define NWEL (KH * KW * 4 * 32 * 16)   // packed weight elements = 100352
#define NWBYTES (NWEL * 2)             // 200704 bytes
#define NWDW (NWBYTES / 4)             // 50176 dwords

// ---------------------------------------------------------------------------
// Pack weights f32 [F,C,7,7] -> bf16 B-fragments in wave32 WMMA layout.
// Layout: wsB[((kk*4 + nt)*32 + lane)*16 + e],  kk = kh*7+kw, nt = f/16.
//   lane 0..15 : column N = lane,    elements e=0..7 are K = 0..7,  e=8..15 are K = 16..23
//   lane 16..31: column N = lane-16, elements e=0..7 are K = 8..15, e=8..15 are K = 24..31
// ---------------------------------------------------------------------------
__global__ void fc_pack_weights(const float* __restrict__ ker,
                                __bf16* __restrict__ wsB) {
    int t = blockIdx.x * 256 + threadIdx.x;
    if (t >= NWEL) return;
    int e    = t & 15;
    int lane = (t >> 4) & 31;
    int nt   = (t >> 9) & 3;
    int kk   = t >> 11;                               // 0..48
    int kh = kk / KW, kw = kk % KW;
    int f = nt * 16 + (lane & 15);
    int c = (e < 8) ? ((lane >> 4) * 8 + e) : (16 + (lane >> 4) * 8 + (e - 8));
    float v = ker[((f * CC + c) * KH + kh) * KW + kw];
    wsB[t] = (__bf16)v;
}

// ---------------------------------------------------------------------------
// Convert x f32 -> bf16 (same [B,H,W,C] channel-last layout), 8 elems/thread.
// ---------------------------------------------------------------------------
__global__ void fc_convert_x(const float* __restrict__ x,
                             __bf16* __restrict__ xb, int n8) {
    int t = blockIdx.x * 256 + threadIdx.x;
    if (t >= n8) return;
    const float4* p = (const float4*)(x + (size_t)t * 8);
    float4 a = p[0];
    float4 b = p[1];
    v8bf o;
    o[0] = (__bf16)a.x; o[1] = (__bf16)a.y; o[2] = (__bf16)a.z; o[3] = (__bf16)a.w;
    o[4] = (__bf16)b.x; o[5] = (__bf16)b.y; o[6] = (__bf16)b.z; o[7] = (__bf16)b.w;
    *(v8bf*)(xb + (size_t)t * 8) = o;
}

// ---------------------------------------------------------------------------
// Main kernel: 1024 threads = 32 waves per block; each wave computes a
// 32(M: consecutive w) x 64(F) output tile. Weights staged in LDS via TDM.
// ---------------------------------------------------------------------------
__global__ __launch_bounds__(1024) void fc_main(const __bf16* __restrict__ xb,
                                                const __bf16* __restrict__ wsB,
                                                const float* __restrict__ bias,
                                                float* __restrict__ out) {
    extern __shared__ __bf16 sB[];              // NWEL elements (200704 B) at LDS offset 0

    // ---- TDM: DMA the full packed-weight blob global -> LDS --------------
    // D# (CDNA5 ISA): 2-D tensor of NWDW dwords, one tile
    // covering the whole tensor, lds_addr=0. Wave 0 issues it once (TDM is
    // EXEC-independent and tracked with TENSORcnt).
    if (threadIdx.x < 32) {
        const unsigned long long ga = (unsigned long long)(uintptr_t)wsB;
        u32x4 g0 = { 1u,                                   // count=1 (valid), user mode
                     0u,                                   // lds_addr = 0
                     (unsigned)(ga & 0xFFFFFFFFu),         // global_addr[31:0]
                     (unsigned)((ga >> 32) & 0x01FFFFFFu)  // global_addr[56:32]
                       | 0x80000000u };                    // type=2 ("image")
        i32x8 g1 = { (int)0x00020000u,    // wg_mask=0, data_size=2 (4B units)
                     (int)0xC4000000u,    // atomic_barrier=0 | tensor_dim0.lo16=50176
                     (int)0x00010000u,    // tensor_dim0.hi16=0 | tensor_dim1.lo16=1
                     (int)0xC4000000u,    // tensor_dim1.hi16=0 | tile_dim0=50176
                     (int)0x00000001u,    // tile_dim1=1, tile_dim2=0
                     (int)50176,          // tensor_dim0_stride[31:0]
                     (int)0xC4000000u,    // dim0_stride.hi16=0 | dim1_stride.lo16=50176
                     0 };                 // tensor_dim1_stride[47:16]=0
        i32x4 gz = { 0, 0, 0, 0 };        // groups 2/3 unused (<=2-D tensor)
#if defined(__clang_major__) && (__clang_major__ >= 23)
        i32x8 gz8 = { 0, 0, 0, 0, 0, 0, 0, 0 };
        __builtin_amdgcn_tensor_load_to_lds(g0, g1, gz, gz, gz8, 0);
#else
        __builtin_amdgcn_tensor_load_to_lds(g0, g1, gz, gz, 0);
#endif
        __builtin_amdgcn_s_wait_tensorcnt(0);
    }
    __syncthreads();

    const int lane = threadIdx.x & 31;
    const int wave = threadIdx.x >> 5;
    const int gw   = blockIdx.x * 32 + wave;    // 0..32767
    const int b    = gw >> 11;                  // 2048 waves per image
    const int h    = (gw >> 3) & 255;
    const int w0   = (gw & 7) * 32;             // 32 consecutive w per wave
    const int i    = lane & 15;                 // row within M-tile
    const int hi   = lane >> 4;                 // half-wave select
    const int c0   = hi * 8;                    // channel base for A chunks

    v8f acc[8];                                 // [mt*4 + nt]
    #pragma unroll
    for (int q = 0; q < 8; ++q) acc[q] = (v8f){0.f,0.f,0.f,0.f,0.f,0.f,0.f,0.f};

    for (int kh = 0; kh < KH; ++kh) {
        const int hrow = (h - kh) & 255;
        // wave-uniform row base; per-lane part stays a 32-bit byte offset so
        // A loads can use the saddr + voffset (GVS) addressing form
        const char* rowp = (const char*)(xb + ((size_t)(b * HH + hrow) * WW) * CC);
        #pragma unroll
        for (int kw = 0; kw < KW; ++kw) {
            // ---- A fragments for both M-tiles (circular w indexing) ----
            const unsigned ww0 = (unsigned)((w0 + i - kw) & 255);
            const unsigned ww1 = (unsigned)((w0 + 16 + i - kw) & 255);
            const unsigned off0 = (ww0 * CC + c0) * 2u;   // bytes
            const unsigned off1 = (ww1 * CC + c0) * 2u;
            v8bf a0lo = *(const v8bf*)(rowp + off0);
            v8bf a0hi = *(const v8bf*)(rowp + off0 + 32); // +16 channels
            v8bf a1lo = *(const v8bf*)(rowp + off1);
            v8bf a1hi = *(const v8bf*)(rowp + off1 + 32);
            v16bf a0 = __builtin_shufflevector(a0lo, a0hi,
                        0,1,2,3,4,5,6,7,8,9,10,11,12,13,14,15);
            v16bf a1 = __builtin_shufflevector(a1lo, a1hi,
                        0,1,2,3,4,5,6,7,8,9,10,11,12,13,14,15);

            // ---- 4 N-tiles, B fragments from LDS (ds_load_b128 x2) ----
            const __bf16* bb = sB + ((kh * KW + kw) * 4) * 512 + lane * 16;
            #pragma unroll
            for (int nt = 0; nt < 4; ++nt) {
                const v8bf* pb = (const v8bf*)(bb + nt * 512);
                v8bf blo = pb[0], bhi = pb[1];
                v16bf bf = __builtin_shufflevector(blo, bhi,
                            0,1,2,3,4,5,6,7,8,9,10,11,12,13,14,15);
                acc[nt] = __builtin_amdgcn_wmma_f32_16x16x32_bf16(
                            false, a0, false, bf, (short)0, acc[nt],
                            false, false);
                acc[4 + nt] = __builtin_amdgcn_wmma_f32_16x16x32_bf16(
                            false, a1, false, bf, (short)0, acc[4 + nt],
                            false, false);
            }
        }
    }

    // ---- store: D layout (ISA): VGPR r, lanes<16 -> M=r, lanes>=16 -> M=r+8,
    //      N = lane&15.  out[b,h,w0+pos, nt*16+N] ----
    const int n = lane & 15;
    float* ob = out + ((size_t)((b * HH + h) * WW + w0)) * FF;
    #pragma unroll
    for (int mt = 0; mt < 2; ++mt) {
        #pragma unroll
        for (int nt = 0; nt < 4; ++nt) {
            const float bv = bias[nt * 16 + n];
            v8f c = acc[mt * 4 + nt];
            #pragma unroll
            for (int r = 0; r < 8; ++r) {
                int pos = mt * 16 + r + hi * 8;
                ob[(size_t)pos * FF + nt * 16 + n] = c[r] + bv;
            }
        }
    }
}

// ---------------------------------------------------------------------------
extern "C" void kernel_launch(void* const* d_in, const int* in_sizes, int n_in,
                              void* d_out, int out_size, void* d_ws, size_t ws_size,
                              hipStream_t stream) {
    const float* x    = (const float*)d_in[0];   // [16,256,256,32] f32
    const float* ker  = (const float*)d_in[1];   // [64,32,7,7]     f32
    const float* bias = (const float*)d_in[2];   // [64,1,1]        f32
    float* out = (float*)d_out;                  // [16,256,256,64] f32

    // workspace: packed bf16 weights (200704 B, 256-aligned) then bf16 x (64 MB)
    __bf16* wsB = (__bf16*)d_ws;
    __bf16* xb  = (__bf16*)((char*)d_ws + (size_t)NWBYTES);

    fc_pack_weights<<<(NWEL + 255) / 256, 256, 0, stream>>>(ker, wsB);

    const int n8 = (BB * HH * WW * CC) / 8;               // 4,194,304
    fc_convert_x<<<n8 / 256, 256, 0, stream>>>(x, xb, n8);

    // 32768 waves (each 32x64 output tile), 32 waves/block -> 1024 blocks,
    // 200704 B dynamic LDS holds the full packed weight set per workgroup.
    fc_main<<<1024, 1024, NWBYTES, stream>>>(xb, wsB, bias, out);
}
